// QNetwork_19224273617469
// MI455X (gfx1250) — compile-verified, matrix operand order checked by reading, not run on hardware
//
#include <hip/hip_runtime.h>
#include <stdint.h>

// ---------------------------------------------------------------------------
// CDNA5 / gfx1250 bf16 WMMA types and helpers
// ---------------------------------------------------------------------------
typedef __attribute__((ext_vector_type(16))) __bf16 bf16x16;
typedef __attribute__((ext_vector_type(8)))  float  v8f;
typedef __attribute__((ext_vector_type(4)))  int    v4i;

union Frag {
  uint4   q[2];   // two 16B chunks (8 bf16 each)
  bf16x16 v;
};

// ---- async global->LDS path (gfx1250 ASYNCcnt), guarded -------------------
#if defined(__has_builtin)
#if __has_builtin(__builtin_amdgcn_global_load_async_to_lds_b128) && \
    __has_builtin(__builtin_amdgcn_s_wait_asynccnt)
#define USE_ASYNC_LDS 1
#endif
#endif
#ifndef USE_ASYNC_LDS
#define USE_ASYNC_LDS 0
#endif

typedef __attribute__((address_space(1))) v4i* gv4i_p;   // global int4*
typedef __attribute__((address_space(3))) v4i* lv4i_p;   // LDS int4*

__device__ __forceinline__ void async_copy16(const uint16_t* g, uint16_t* l) {
#if USE_ASYNC_LDS
  __builtin_amdgcn_global_load_async_to_lds_b128((gv4i_p)g, (lv4i_p)l, 0, 0);
#endif
}
__device__ __forceinline__ void async_wait0() {
#if USE_ASYNC_LDS
  __builtin_amdgcn_s_wait_asynccnt(0);
#endif
}

__device__ __forceinline__ uint16_t f2bf(float f) {
  union { float f; uint32_t u; } x; x.f = f;
  uint32_t r = x.u + 0x7FFFu + ((x.u >> 16) & 1u);   // round-to-nearest-even
  return (uint16_t)(r >> 16);
}
__device__ __forceinline__ float bf2f(uint16_t h) {
  union { uint32_t u; float f; } x; x.u = ((uint32_t)h) << 16;
  return x.f;
}
__device__ __forceinline__ v8f v8zero() {
  v8f z = {0.f, 0.f, 0.f, 0.f, 0.f, 0.f, 0.f, 0.f};
  return z;
}
__device__ __forceinline__ v8f wmma_bf16(const Frag& a, const Frag& b, v8f c) {
  return __builtin_amdgcn_wmma_f32_16x16x32_bf16(
      false, a.v, false, b.v, (short)0, c, false, false);
}

// Symmetry-branch index map for a (S+1)x(S+1) square kernel.
__device__ __forceinline__ void sym_map(int br, int i, int j, int S, int& oi, int& oj) {
  switch (br) {
    case 0: oi = i;     oj = j;     break;
    case 1: oi = S - i; oj = j;     break;
    case 2: oi = i;     oj = S - j; break;
    case 3: oi = S - i; oj = S - j; break;
    case 4: oi = j;     oj = i;     break;
    case 5: oi = j;     oj = S - i; break;
    case 6: oi = S - j; oj = i;     break;
    default: oi = S - j; oj = S - i; break;
  }
}

// ---------------------------------------------------------------------------
// Weight-preparation kernels (run every call; tiny cost)
// ---------------------------------------------------------------------------
__global__ void prep_conv1(const float* __restrict__ w, const float* __restrict__ b,
                           uint16_t* __restrict__ Wt, float* __restrict__ bias) {
  int idx = blockIdx.x * blockDim.x + threadIdx.x;
  if (idx >= 2048 * 256) return;
  int o = idx >> 8, f = idx & 255;
  int oc = o >> 4, pos = o & 15;
  int br = oc >> 4, k = oc & 15;
  int r = pos >> 2, c = pos & 3;
  int pin = f >> 4, ic = f & 15;
  int ri = pin >> 2, ci = pin & 3;
  int dr = ri - r + 1, dc = ci - c + 1;   // pad=1, 3x3
  float val = 0.f;
  if (dr >= 0 && dr < 3 && dc >= 0 && dc < 3) {
    int oi, oj; sym_map(br, dr, dc, 2, oi, oj);
    val = w[((k * 16 + ic) * 3 + oi) * 3 + oj];
  }
  Wt[(size_t)o * 256 + f] = f2bf(val);
  if (f == 0) bias[o] = b[k];
}

__global__ void prep_conv3(const float* __restrict__ w, const float* __restrict__ b,
                           uint16_t* __restrict__ Wt, float* __restrict__ bias) {
  int idx = blockIdx.x * blockDim.x + threadIdx.x;
  if (idx >= 128 * 2048) return;
  int o3 = idx >> 11, f1 = idx & 2047;
  int br = o3 >> 4, k = o3 & 15;
  int oc1 = f1 >> 4, pos = f1 & 15;
  int r = pos >> 2, c = pos & 3;
  int oi, oj; sym_map(br, r, c, 3, oi, oj);
  float val = w[((k * 128 + oc1) * 4 + oi) * 4 + oj];
  Wt[(size_t)o3 * 2048 + f1] = f2bf(val);
  if (f1 == 0) bias[o3] = b[k];
}

__global__ void prep_rows(const float* __restrict__ w, const float* __restrict__ b,
                          uint16_t* __restrict__ Wt, float* __restrict__ bias) {
  int idx = blockIdx.x * blockDim.x + threadIdx.x;
  if (idx >= 256 * 256) return;
  int o = idx >> 8, f = idx & 255;
  int ch = o >> 2, r = o & 3;
  int blk = ch >> 4, k = ch & 15;
  int pin = f >> 4, ic = f & 15;
  int ri = pin >> 2, ci = pin & 3;
  float val = 0.f;
  if (blk < 2) {               // s1 / s2 (height-1 flip is identity)
    if (ri == r) val = w[(k * 16 + ic) * 4 + ci];
  } else {                     // transposed branch
    if (ci == r) val = w[(k * 16 + ic) * 4 + ri];
  }
  Wt[(size_t)o * 256 + f] = f2bf(val);
  if (f == 0) bias[o] = b[k];
}

__global__ void prep_fc(const float* __restrict__ w, const float* __restrict__ b,
                        const float* __restrict__ bn, int N, int K,
                        uint16_t* __restrict__ Wt, float* __restrict__ bias) {
  int idx = blockIdx.x * blockDim.x + threadIdx.x;
  if (idx >= N * K) return;
  int n = idx / K, k = idx - n * K;
  float s = 1.f, beta = 0.f, m = 0.f;
  if (bn) {
    float g = bn[n], bb = bn[N + n], mm = bn[2 * N + n], vv = bn[3 * N + n];
    s = g * rsqrtf(vv + 1e-5f); beta = bb; m = mm;
  }
  Wt[(size_t)n * K + k] = f2bf(w[(size_t)n * K + k] * s);
  if (k == 0) bias[n] = (b[n] - m) * s + beta;
}

__global__ void embed_kernel(const int* __restrict__ state, const float* __restrict__ emb,
                             uint16_t* __restrict__ A0, int B) {
  int t = blockIdx.x * blockDim.x + threadIdx.x;
  if (t >= B * 16) return;
  int s = t >> 4, p = t & 15;
  int idx = state[t];
  uint16_t* dst = A0 + (size_t)s * 256 + p * 16;
  const float* src = emb + idx * 16;
#pragma unroll
  for (int ch = 0; ch < 16; ++ch) dst[ch] = f2bf(src[ch]);
}

// ---------------------------------------------------------------------------
// Generic tiled bf16 GEMM:  C[M,N] = act(A[M,K] * Wt[N,K]^T + bias)
// block = 256 threads (8 waves); tile 64(M) x 128*NB(N); each wave:
// 4 WMMA M-subtiles x NB 16-col strips (A fragments reused across strips).
// A staged into LDS with a double-buffered async global->LDS pipeline.
// ---------------------------------------------------------------------------
template <int NB>
__global__ __launch_bounds__(256)
void gemm_bf16(const uint16_t* __restrict__ A, const uint16_t* __restrict__ Wt,
               const float* __restrict__ bias, uint16_t* __restrict__ C,
               int K, int ldc, int relu) {
  __shared__ __align__(16) uint16_t As[2][64 * 40];  // 64 rows x 32 K, stride 40

  const int tid  = threadIdx.x;
  const int wave = tid >> 5;
  const int lane = tid & 31;
  const int lm   = lane & 15;
  const int klo  = (lane < 16) ? 0 : 8;              // 16-bit A/B K striping

  const size_t m0 = (size_t)blockIdx.y * 64;
  const int    n0 = blockIdx.x * (128 * NB) + wave * (16 * NB);

  v8f acc[NB][4];
#pragma unroll
  for (int nb = 0; nb < NB; ++nb)
#pragma unroll
    for (int t = 0; t < 4; ++t) acc[nb][t] = v8zero();

  // per-thread 16B staging slice
  const int srow = tid >> 2, sseg = tid & 3;
  const uint16_t* gsrc = A + (m0 + srow) * K + sseg * 8;
  uint16_t* ldst[2] = {&As[0][srow * 40 + sseg * 8], &As[1][srow * 40 + sseg * 8]};

  const int nk = K >> 5;

#if USE_ASYNC_LDS
  async_copy16(gsrc, ldst[0]);
  for (int i = 0; i < nk; ++i) {
    const int cur = i & 1;
    const int k0  = i << 5;
    async_wait0();
    __syncthreads();
    if (i + 1 < nk) async_copy16(gsrc + k0 + 32, ldst[cur ^ 1]);  // overlap compute
#else
  for (int i = 0; i < nk; ++i) {
    const int cur = 0;
    const int k0  = i << 5;
    __syncthreads();
    *(uint4*)ldst[0] = *(const uint4*)(gsrc + k0);
    __syncthreads();
#endif

    Frag bf[NB];
#pragma unroll
    for (int nb = 0; nb < NB; ++nb) {
      const uint16_t* wp = Wt + (size_t)(n0 + nb * 16 + lm) * K + k0 + klo;
      bf[nb].q[0] = *(const uint4*)(wp);
      bf[nb].q[1] = *(const uint4*)(wp + 16);
      __builtin_prefetch(wp + 32, 0, 3);             // global_prefetch_b8
    }

#pragma unroll
    for (int t = 0; t < 4; ++t) {
      Frag af;
      const uint16_t* ap = &As[cur][(t * 16 + lm) * 40 + klo];
      af.q[0] = *(const uint4*)(ap);
      af.q[1] = *(const uint4*)(ap + 16);
#pragma unroll
      for (int nb = 0; nb < NB; ++nb) acc[nb][t] = wmma_bf16(af, bf[nb], acc[nb][t]);
    }
  }

#pragma unroll
  for (int nb = 0; nb < NB; ++nb) {
    const int coln = n0 + nb * 16 + lm;
    const float bv = bias[coln];
#pragma unroll
    for (int t = 0; t < 4; ++t) {
#pragma unroll
      for (int i = 0; i < 8; ++i) {
        int rl = t * 16 + ((lane < 16) ? i : i + 8);
        float v = acc[nb][t][i] + bv;
        if (relu) v = v > 0.f ? v : 0.f;
        C[(m0 + rl) * (size_t)ldc + coln] = f2bf(v);
      }
    }
  }
}

// ---------------------------------------------------------------------------
// Fused cc1 -> ReLU -> cc3 -> ReLU (avoids the 128MB cc1 intermediate).
// ---------------------------------------------------------------------------
__global__ __launch_bounds__(256)
void fused_conv13(const uint16_t* __restrict__ A0, const uint16_t* __restrict__ Wc1t,
                  const float* __restrict__ bc1, const uint16_t* __restrict__ Wc3t,
                  const float* __restrict__ bc3, uint16_t* __restrict__ X, int ldx) {
  __shared__ __align__(16) uint16_t A0s[64 * 264];   // 64 x 256, stride 264
  __shared__ __align__(16) uint16_t C1s[64 * 136];   // 64 x 128, stride 136

  const int tid  = threadIdx.x;
  const int wave = tid >> 5;
  const int lane = tid & 31;
  const int lm   = lane & 15;
  const int klo  = (lane < 16) ? 0 : 8;
  const size_t m0 = (size_t)blockIdx.x * 64;

  // stage full A0 tile (64 x 256 bf16 = 32KB)
#if USE_ASYNC_LDS
  for (int e = tid; e < 64 * 32; e += 256)
    async_copy16(A0 + (m0 + (e >> 5)) * 256 + (e & 31) * 8,
                 &A0s[(e >> 5) * 264 + (e & 31) * 8]);
  async_wait0();
#else
  for (int e = tid; e < 64 * 32; e += 256) {
    int row = e >> 5, seg = e & 31;
    *(uint4*)&A0s[row * 264 + seg * 8] =
        *(const uint4*)(A0 + (m0 + row) * 256 + seg * 8);
  }
#endif
  __syncthreads();

  v8f acc3[4] = {v8zero(), v8zero(), v8zero(), v8zero()};
  const int n3 = wave * 16 + lm;

  for (int chunk = 0; chunk < 16; ++chunk) {
    const int c1n = chunk * 128 + wave * 16 + lm;
    v8f acc1[4] = {v8zero(), v8zero(), v8zero(), v8zero()};

    for (int k0 = 0; k0 < 256; k0 += 32) {
      Frag bf;
      const uint16_t* wp = Wc1t + (size_t)c1n * 256 + k0 + klo;
      bf.q[0] = *(const uint4*)(wp);
      bf.q[1] = *(const uint4*)(wp + 16);
      __builtin_prefetch(wp + 32, 0, 3);
#pragma unroll
      for (int t = 0; t < 4; ++t) {
        Frag af;
        const uint16_t* ap = &A0s[(t * 16 + lm) * 264 + k0 + klo];
        af.q[0] = *(const uint4*)(ap);
        af.q[1] = *(const uint4*)(ap + 16);
        acc1[t] = wmma_bf16(af, bf, acc1[t]);
      }
    }

    const float b1 = bc1[c1n];
    __syncthreads();       // previous chunk's cc3 reads of C1s complete
#pragma unroll
    for (int t = 0; t < 4; ++t) {
#pragma unroll
      for (int i = 0; i < 8; ++i) {
        int rl = t * 16 + ((lane < 16) ? i : i + 8);
        float v = acc1[t][i] + b1;
        v = v > 0.f ? v : 0.f;
        C1s[rl * 136 + wave * 16 + lm] = f2bf(v);
      }
    }
    __syncthreads();

    for (int kk = 0; kk < 128; kk += 32) {
      Frag bf;
      const uint16_t* wp = Wc3t + (size_t)n3 * 2048 + chunk * 128 + kk + klo;
      bf.q[0] = *(const uint4*)(wp);
      bf.q[1] = *(const uint4*)(wp + 16);
#pragma unroll
      for (int t = 0; t < 4; ++t) {
        Frag af;
        const uint16_t* ap = &C1s[(t * 16 + lm) * 136 + kk + klo];
        af.q[0] = *(const uint4*)(ap);
        af.q[1] = *(const uint4*)(ap + 16);
        acc3[t] = wmma_bf16(af, bf, acc3[t]);
      }
    }
  }

  const float b3 = bc3[n3];
#pragma unroll
  for (int t = 0; t < 4; ++t) {
#pragma unroll
    for (int i = 0; i < 8; ++i) {
      int rl = t * 16 + ((lane < 16) ? i : i + 8);
      float v = acc3[t][i] + b3;
      v = v > 0.f ? v : 0.f;
      X[(m0 + rl) * (size_t)ldx + n3] = f2bf(v);
    }
  }
}

// ---------------------------------------------------------------------------
// Head: v3/a3 + dueling combine. One thread per sample.
// ---------------------------------------------------------------------------
__global__ void head_kernel(const uint16_t* __restrict__ V2, const uint16_t* __restrict__ A2,
                            const float* __restrict__ v3w, const float* __restrict__ v3b,
                            const float* __restrict__ a3w, const float* __restrict__ a3b,
                            float* __restrict__ out, int B) {
  int s = blockIdx.x * blockDim.x + threadIdx.x;
  if (s >= B) return;
  const uint16_t* vp = V2 + (size_t)s * 256;
  const uint16_t* ap = A2 + (size_t)s * 256;
  float v = v3b[0];
  float a0 = a3b[0], a1 = a3b[1], a2 = a3b[2], a3 = a3b[3];
  for (int j = 0; j < 256; ++j) {
    float xv = bf2f(vp[j]);
    float xa = bf2f(ap[j]);
    v  += v3w[j] * xv;
    a0 += a3w[j] * xa;
    a1 += a3w[256 + j] * xa;
    a2 += a3w[512 + j] * xa;
    a3 += a3w[768 + j] * xa;
  }
  float mean = 0.25f * (a0 + a1 + a2 + a3);
  out[(size_t)s * 4 + 0] = v + a0 - mean;
  out[(size_t)s * 4 + 1] = v + a1 - mean;
  out[(size_t)s * 4 + 2] = v + a2 - mean;
  out[(size_t)s * 4 + 3] = v + a3 - mean;
}

// ---------------------------------------------------------------------------
// Host launcher
// ---------------------------------------------------------------------------
extern "C" void kernel_launch(void* const* d_in, const int* in_sizes, int n_in,
                              void* d_out, int out_size, void* d_ws, size_t ws_size,
                              hipStream_t stream) {
  const int B = in_sizes[0] / 16;

  const int*   state = (const int*)  d_in[0];
  const float* emb_w = (const float*)d_in[1];
  const float* c1_w  = (const float*)d_in[2];
  const float* c1_b  = (const float*)d_in[3];
  const float* c3_w  = (const float*)d_in[4];
  const float* c3_b  = (const float*)d_in[5];
  const float* cr_w  = (const float*)d_in[6];
  const float* cr_b  = (const float*)d_in[7];
  const float* fc1_w = (const float*)d_in[8];
  const float* fc1_b = (const float*)d_in[9];
  const float* fc1_bn= (const float*)d_in[10];
  const float* v1_w  = (const float*)d_in[11];
  const float* v1_b  = (const float*)d_in[12];
  const float* v1_bn = (const float*)d_in[13];
  const float* v2_w  = (const float*)d_in[14];
  const float* v2_b  = (const float*)d_in[15];
  const float* v2_bn = (const float*)d_in[16];
  const float* v3_w  = (const float*)d_in[17];
  const float* v3_b  = (const float*)d_in[18];
  const float* a1_w  = (const float*)d_in[19];
  const float* a1_b  = (const float*)d_in[20];
  const float* a1_bn = (const float*)d_in[21];
  const float* a2_w  = (const float*)d_in[22];
  const float* a2_b  = (const float*)d_in[23];
  const float* a2_bn = (const float*)d_in[24];
  const float* a3_w  = (const float*)d_in[25];
  const float* a3_b  = (const float*)d_in[26];

  char* ws = (char*)d_ws;
  size_t off = 0;
  auto alloc = [&](size_t bytes) -> char* {
    char* p = ws + off;
    off = (off + bytes + 255) & ~(size_t)255;
    return p;
  };

  uint16_t* Wc1t  = (uint16_t*)alloc(2048 * 256 * 2);
  float*    bc1   = (float*)   alloc(2048 * 4);
  uint16_t* Wc3t  = (uint16_t*)alloc(128 * 2048 * 2);
  float*    bc3   = (float*)   alloc(128 * 4);
  uint16_t* Wrt   = (uint16_t*)alloc(256 * 256 * 2);
  float*    brr   = (float*)   alloc(256 * 4);
  uint16_t* Wfc1t = (uint16_t*)alloc(384 * 384 * 2);
  float*    bfc1  = (float*)   alloc(384 * 4);
  uint16_t* Wv1t  = (uint16_t*)alloc(256 * 384 * 2);
  float*    bv1   = (float*)   alloc(256 * 4);
  uint16_t* Wa1t  = (uint16_t*)alloc(256 * 384 * 2);
  float*    ba1   = (float*)   alloc(256 * 4);
  uint16_t* Wv2t  = (uint16_t*)alloc(256 * 256 * 2);
  float*    bv2   = (float*)   alloc(256 * 4);
  uint16_t* Wa2t  = (uint16_t*)alloc(256 * 256 * 2);
  float*    ba2   = (float*)   alloc(256 * 4);

  uint16_t* A0 = (uint16_t*)alloc((size_t)B * 256 * 2);
  uint16_t* X  = (uint16_t*)alloc((size_t)B * 384 * 2);
  uint16_t* F1 = (uint16_t*)alloc((size_t)B * 384 * 2);
  uint16_t* V  = (uint16_t*)alloc((size_t)B * 256 * 2);
  uint16_t* Aa = (uint16_t*)alloc((size_t)B * 256 * 2);
  uint16_t* V2 = (uint16_t*)alloc((size_t)B * 256 * 2);
  uint16_t* A2 = (uint16_t*)alloc((size_t)B * 256 * 2);

  // --- weight prep (stateless; every call) ---
  prep_conv1<<<(2048 * 256) / 256, 256, 0, stream>>>(c1_w, c1_b, Wc1t, bc1);
  prep_conv3<<<(128 * 2048) / 256, 256, 0, stream>>>(c3_w, c3_b, Wc3t, bc3);
  prep_rows <<<(256 * 256) / 256, 256, 0, stream>>>(cr_w, cr_b, Wrt, brr);
  prep_fc<<<(384 * 384) / 256, 256, 0, stream>>>(fc1_w, fc1_b, fc1_bn, 384, 384, Wfc1t, bfc1);
  prep_fc<<<(256 * 384) / 256, 256, 0, stream>>>(v1_w, v1_b, v1_bn, 256, 384, Wv1t, bv1);
  prep_fc<<<(256 * 384) / 256, 256, 0, stream>>>(a1_w, a1_b, a1_bn, 256, 384, Wa1t, ba1);
  prep_fc<<<(256 * 256) / 256, 256, 0, stream>>>(v2_w, v2_b, v2_bn, 256, 256, Wv2t, bv2);
  prep_fc<<<(256 * 256) / 256, 256, 0, stream>>>(a2_w, a2_b, a2_bn, 256, 256, Wa2t, ba2);

  // --- activations ---
  embed_kernel<<<(B * 16) / 256, 256, 0, stream>>>(state, emb_w, A0, B);

  const int MB = B / 64;   // M tiles of 64 rows

  // fused cc1->cc3 -> X[:, 0:128]
  fused_conv13<<<MB, 256, 0, stream>>>(A0, Wc1t, bc1, Wc3t, bc3, X, 384);
  // row-conv -> X[:, 128:384]           (N=256, wide tiles)
  gemm_bf16<2><<<dim3(1, MB), 256, 0, stream>>>(A0, Wrt, brr, X + 128, 256, 384, 1);
  // fc1 -> F1                            (N=384, narrow tiles)
  gemm_bf16<1><<<dim3(3, MB), 256, 0, stream>>>(X, Wfc1t, bfc1, F1, 384, 384, 1);
  // value / advantage hidden layers      (N=256, wide tiles)
  gemm_bf16<2><<<dim3(1, MB), 256, 0, stream>>>(F1, Wv1t, bv1, V, 384, 256, 1);
  gemm_bf16<2><<<dim3(1, MB), 256, 0, stream>>>(F1, Wa1t, ba1, Aa, 384, 256, 1);
  gemm_bf16<2><<<dim3(1, MB), 256, 0, stream>>>(V, Wv2t, bv2, V2, 256, 256, 1);
  gemm_bf16<2><<<dim3(1, MB), 256, 0, stream>>>(Aa, Wa2t, ba2, A2, 256, 256, 1);

  // dueling head
  head_kernel<<<(B + 255) / 256, 256, 0, stream>>>(V2, A2, v3_w, v3_b, a3_w, a3_b,
                                                   (float*)d_out, B);
}
